// AurexSupplyChainGNN_8778913153100
// MI455X (gfx1250) — matrix operand order repaired and checked by reading, not compiled, not run on hardware
//
#include <hip/hip_runtime.h>
#include <hip/hip_bf16.h>
#include <math.h>

typedef __bf16 bf16_t;
typedef __attribute__((ext_vector_type(16))) __bf16 v16bf;
typedef __attribute__((ext_vector_type(8)))  __bf16 v8bf;
typedef __attribute__((ext_vector_type(8)))  float  v8f;

#define FIN    128
#define HD     512      // HEADS*HID
#define HEADS  8
#define HIDC   64
#define NCLS   10
#define NGRAPH 16

// ---------------------------------------------------------------- utilities
__global__ void k_fill_f32(float* __restrict__ p, float v, int n) {
  int i = blockIdx.x * blockDim.x + threadIdx.x;
  if (i < n) p[i] = v;
}

__global__ void k_cvt_bf16(const float* __restrict__ in, bf16_t* __restrict__ out, int n) {
  int i = blockIdx.x * blockDim.x + threadIdx.x;
  if (i < n) out[i] = (bf16_t)in[i];
}

// W3 is [10,512]; pad to [16,512] bf16 with zero rows so the WMMA N-tile is full.
__global__ void k_cvt_w3(const float* __restrict__ in, bf16_t* __restrict__ out) {
  int i = blockIdx.x * blockDim.x + threadIdx.x;
  if (i < 16 * HD) {
    int r = i / HD;
    out[i] = (r < NCLS) ? (bf16_t)in[i] : (bf16_t)0.0f;
  }
}

__device__ __forceinline__ void atomicMaxF(float* addr, float v) {
  // float max via integer atomics (works for mixed signs, init to very negative)
  if (v >= 0.0f) atomicMax((int*)addr, __float_as_int(v));
  else           atomicMin((unsigned int*)addr, __float_as_uint(v));
}

// ------------------------------------------------------------- WMMA GEMM
// C[M,Nc] = A[M,K] @ B[Nc,K]^T ; A,B bf16 row-major, C f32 row-major.
// One wave computes a 16 x (16*NT) strip using v_wmma_f32_16x16x32_bf16.
template<int NT>
__global__ void k_gemm_bf16(const bf16_t* __restrict__ A,
                            const bf16_t* __restrict__ B,
                            float* __restrict__ C,
                            int M, int Nc, int K) {
  const int wave = (blockIdx.x * blockDim.x + threadIdx.x) >> 5;
  const int lane = threadIdx.x & 31;
  const int ntg  = Nc / (16 * NT);
  const int mt   = wave / ntg;
  const int ng   = wave % ntg;
  if (mt * 16 >= M) return;
  const int m0   = mt * 16;
  const int n0   = ng * 16 * NT;
  const int half = lane >> 4;     // K-half select: contiguous k = 16*half + s for A and B
  const int l15  = lane & 15;

  v8f acc[NT];
  for (int t = 0; t < NT; ++t) acc[t] = v8f{0,0,0,0,0,0,0,0};

  const bf16_t* arow = A + (size_t)(m0 + l15) * K + half * 16;
  for (int k0 = 0; k0 < K; k0 += 32) {
    // A 16x32 fragment: lanes 0-15 rows, elems 0-15 = K[k0..k0+15]
    // (lanes 16-31 take K[k0+16..k0+31]) -> two 16B contiguous loads.
    v8bf a_lo = *(const v8bf*)(arow + k0);
    v8bf a_hi = *(const v8bf*)(arow + k0 + 8);
    v16bf a = __builtin_shufflevector(a_lo, a_hi, 0,1,2,3,4,5,6,7,8,9,10,11,12,13,14,15);
#pragma unroll
    for (int t = 0; t < NT; ++t) {
      // B 32x16 fragment: lane holds column n0+l15, 16 consecutive K values (32B load)
      const bf16_t* bp = B + (size_t)(n0 + t * 16 + l15) * K + k0 + half * 16;
      v16bf b = *(const v16bf*)bp;
      acc[t] = __builtin_amdgcn_wmma_f32_16x16x32_bf16(
          false, a, false, b, (short)0, acc[t], false, false);
    }
  }
#pragma unroll
  for (int t = 0; t < NT; ++t) {
    float* cp = C + (size_t)(m0 + half * 8) * Nc + n0 + t * 16 + l15;
#pragma unroll
    for (int j = 0; j < 8; ++j) cp[(size_t)j * Nc] = acc[t][j];
  }
}

// ------------------------------------------------- per-node attention scores
__global__ void k_node_scores(const float* __restrict__ h,
                              const float* __restrict__ a_s,
                              const float* __restrict__ a_d,
                              float* __restrict__ es, float* __restrict__ ed,
                              int n, int heads, int hc, int ldh) {
  int i = blockIdx.x * blockDim.x + threadIdx.x;
  if (i >= n * heads) return;
  int node = i / heads, hh = i % heads;
  const float* hp = h + (size_t)node * ldh + hh * hc;
  const float* as = a_s + hh * hc;
  const float* ad = a_d + hh * hc;
  float s = 0.f, d = 0.f;
  for (int c = 0; c < hc; ++c) { float v = hp[c]; s += v * as[c]; d += v * ad[c]; }
  es[i] = s; ed[i] = d;
}

// ----------------------------------------------------------- edge passes
__device__ __forceinline__ void edge_sd(const int* ei, int E, int e, int& s, int& d) {
  if (e < E) { s = ei[e]; d = ei[E + e]; } else { s = d = e - E; }  // self loops
}

__global__ void k_edge_alpha_max(const int* __restrict__ ei, int E, int n, int heads,
                                 const float* __restrict__ es, const float* __restrict__ ed,
                                 float* __restrict__ alpha, float* __restrict__ amax) {
  int i = blockIdx.x * blockDim.x + threadIdx.x;
  if (i >= (E + n) * heads) return;
  int e = i / heads, hh = i % heads, s, d;
  edge_sd(ei, E, e, s, d);
  float a = es[s * heads + hh] + ed[d * heads + hh];
  a = (a > 0.f) ? a : 0.2f * a;     // leaky relu
  alpha[i] = a;
  atomicMaxF(&amax[d * heads + hh], a);
}

__global__ void k_edge_exp_sum(const int* __restrict__ ei, int E, int n, int heads,
                               float* __restrict__ alpha, const float* __restrict__ amax,
                               float* __restrict__ den) {
  int i = blockIdx.x * blockDim.x + threadIdx.x;
  if (i >= (E + n) * heads) return;
  int e = i / heads, hh = i % heads, s, d;
  edge_sd(ei, E, e, s, d);
  float ex = __expf(alpha[i] - amax[d * heads + hh]);
  alpha[i] = ex;
  atomicAdd(&den[d * heads + hh], ex);
}

__global__ void k_edge_scatter(const int* __restrict__ ei, int E, int n, int heads,
                               int hc, int ldh, int ldo,
                               const float* __restrict__ alpha, const float* __restrict__ den,
                               const float* __restrict__ h, float* __restrict__ out) {
  int i = blockIdx.x * blockDim.x + threadIdx.x;
  if (i >= (E + n) * heads) return;
  int e = i / heads, hh = i % heads, s, d;
  edge_sd(ei, E, e, s, d);
  float w = alpha[i] / den[d * heads + hh];
  const float* hp = h + (size_t)s * ldh + hh * hc;
  float* op = out + (size_t)d * ldo + hh * hc;
  for (int c = 0; c < hc; ++c) atomicAdd(&op[c], hp[c] * w);
}

// ------------------------------------------------------------- batch norm
__global__ void k_bn_stats(const float* __restrict__ x, const float* __restrict__ bias,
                           float* __restrict__ sum, float* __restrict__ sumsq,
                           int n, int rows_per_block) {
  int feat = threadIdx.x;                    // blockDim.x == HD
  int r0 = blockIdx.x * rows_per_block;
  int r1 = r0 + rows_per_block; if (r1 > n) r1 = n;
  float b = bias[feat], s = 0.f, q = 0.f;
  for (int r = r0; r < r1; ++r) {
    float v = x[(size_t)r * HD + feat] + b;
    s += v; q += v * v;
  }
  atomicAdd(&sum[feat], s);
  atomicAdd(&sumsq[feat], q);
}

__global__ void k_bn_apply(const float* __restrict__ x, const float* __restrict__ bias,
                           const float* __restrict__ sum, const float* __restrict__ sumsq,
                           const float* __restrict__ gamma, const float* __restrict__ beta,
                           bf16_t* __restrict__ y, int n) {
  int i = blockIdx.x * blockDim.x + threadIdx.x;
  if (i >= n * HD) return;
  int feat = i % HD;
  float invN = 1.0f / (float)n;
  float mu = sum[feat] * invN;
  float var = sumsq[feat] * invN - mu * mu;
  float v = (x[i] + bias[feat] - mu) * rsqrtf(var + 1e-5f) * gamma[feat] + beta[feat];
  y[i] = (bf16_t)(v > 0.f ? v : 0.f);        // relu, bf16 for next WMMA GEMM
}

// ------------------------------------------------------------- pooling + MLP
__global__ void k_pool_acc(const float* __restrict__ h3, const float* __restrict__ b3,
                           const int* __restrict__ batch,
                           float* __restrict__ gsum, float* __restrict__ gmax,
                           float* __restrict__ gcnt, int n) {
  int i = blockIdx.x * blockDim.x + threadIdx.x;
  if (i >= n) return;
  int g = batch[i];
  atomicAdd(&gcnt[g], 1.0f);
  for (int c = 0; c < NCLS; ++c) {
    float v = h3[(size_t)i * NCLS + c] + b3[c];
    atomicAdd(&gsum[g * NCLS + c], v);
    atomicMaxF(&gmax[g * NCLS + c], v);
  }
}

__global__ void k_head_mlp(const float* __restrict__ gsum, const float* __restrict__ gmax,
                           const float* __restrict__ gcnt,
                           const float* __restrict__ Wp1, const float* __restrict__ bp1,
                           const float* __restrict__ Wp2, const float* __restrict__ bp2,
                           float* __restrict__ out) {
  int g = threadIdx.x;
  if (g >= NGRAPH) return;
  float p[2 * NCLS];
  float inv = 1.0f / gcnt[g];
  for (int c = 0; c < NCLS; ++c) {
    p[c] = gsum[g * NCLS + c] * inv;
    p[NCLS + c] = gmax[g * NCLS + c];
  }
  float hid[HIDC];
  for (int k = 0; k < HIDC; ++k) {
    float a = bp1[k];
    for (int j = 0; j < 2 * NCLS; ++j) a += Wp1[k * 2 * NCLS + j] * p[j];
    hid[k] = a > 0.f ? a : 0.f;
  }
  float logit[NCLS]; float mx = -1e30f;
  for (int c = 0; c < NCLS; ++c) {
    float a = bp2[c];
    for (int k = 0; k < HIDC; ++k) a += Wp2[c * HIDC + k] * hid[k];
    logit[c] = a; mx = fmaxf(mx, a);
  }
  float se = 0.f;
  for (int c = 0; c < NCLS; ++c) se += __expf(logit[c] - mx);
  float lse = mx + __logf(se);
  for (int c = 0; c < NCLS; ++c) out[g * NCLS + c] = logit[c] - lse;
}

// ----------------------------------------------------------------- driver
extern "C" void kernel_launch(void* const* d_in, const int* in_sizes, int n_in,
                              void* d_out, int out_size, void* d_ws, size_t ws_size,
                              hipStream_t stream) {
  const float* x    = (const float*)d_in[0];
  const int*   ei   = (const int*)  d_in[1];
  const int*   batch= (const int*)  d_in[2];
  const float* W1   = (const float*)d_in[3];
  const float* a1s  = (const float*)d_in[4];
  const float* a1d  = (const float*)d_in[5];
  const float* b1   = (const float*)d_in[6];
  const float* g1   = (const float*)d_in[7];
  const float* be1  = (const float*)d_in[8];
  const float* W2   = (const float*)d_in[9];
  const float* a2s  = (const float*)d_in[10];
  const float* a2d  = (const float*)d_in[11];
  const float* b2   = (const float*)d_in[12];
  const float* g2   = (const float*)d_in[13];
  const float* be2  = (const float*)d_in[14];
  const float* W3   = (const float*)d_in[15];
  const float* a3s  = (const float*)d_in[16];
  const float* a3d  = (const float*)d_in[17];
  const float* b3   = (const float*)d_in[18];
  const float* Wp1  = (const float*)d_in[19];
  const float* bp1  = (const float*)d_in[20];
  const float* Wp2  = (const float*)d_in[21];
  const float* bp2  = (const float*)d_in[22];

  const int N  = in_sizes[0] / FIN;    // 20000 (multiple of 16)
  const int E  = in_sizes[1] / 2;      // 320000
  const int Ep = E + N;                // edges incl. self-loops

  // ---- workspace bump allocator (256B aligned)
  char* ws = (char*)d_ws; size_t off = 0;
  auto alloc = [&](size_t bytes) -> void* {
    off = (off + 255) & ~(size_t)255;
    void* p = ws + off; off += bytes; return p;
  };
  bf16_t* xb   = (bf16_t*)alloc((size_t)N * FIN * 2);
  bf16_t* w1b  = (bf16_t*)alloc((size_t)HD * FIN * 2);
  bf16_t* w2b  = (bf16_t*)alloc((size_t)HD * HD * 2);
  bf16_t* w3b  = (bf16_t*)alloc((size_t)16 * HD * 2);
  float*  h    = (float*)alloc((size_t)N * HD * 4);
  float*  agg  = (float*)alloc((size_t)N * HD * 4);
  bf16_t* yb   = (bf16_t*)alloc((size_t)N * HD * 2);
  float*  es   = (float*)alloc((size_t)N * HEADS * 4);
  float*  ed   = (float*)alloc((size_t)N * HEADS * 4);
  float*  amax = (float*)alloc((size_t)N * HEADS * 4);
  float*  den  = (float*)alloc((size_t)N * HEADS * 4);
  float*  alpha= (float*)alloc((size_t)Ep * HEADS * 4);
  float*  bnsum= (float*)alloc((size_t)HD * 4);
  float*  bnsq = (float*)alloc((size_t)HD * 4);
  float*  h3   = (float*)alloc((size_t)N * 16 * 4);
  float*  out3 = (float*)alloc((size_t)N * NCLS * 4);
  float*  gsum = (float*)alloc((size_t)NGRAPH * NCLS * 4);
  float*  gmax = (float*)alloc((size_t)NGRAPH * NCLS * 4);
  float*  gcnt = (float*)alloc((size_t)NGRAPH * 4);

  auto cdiv = [](int a, int b) { return (a + b - 1) / b; };
  const int TPB = 256;
  const float NEG = -1e30f;

  // ---- bf16 conversions
  k_cvt_bf16<<<cdiv(N * FIN, TPB), TPB, 0, stream>>>(x,  xb,  N * FIN);
  k_cvt_bf16<<<cdiv(HD * FIN, TPB), TPB, 0, stream>>>(W1, w1b, HD * FIN);
  k_cvt_bf16<<<cdiv(HD * HD,  TPB), TPB, 0, stream>>>(W2, w2b, HD * HD);
  k_cvt_w3 <<<cdiv(16 * HD,  TPB), TPB, 0, stream>>>(W3, w3b);

  const int EH  = Ep * HEADS;
  const int Mt  = cdiv(N, 16);
  const int gemmBlocks4 = cdiv(Mt * (HD / 64), 8);   // NT=4 -> 8 N-tile groups

  // ========================= GAT layer 1 ==========================
  k_gemm_bf16<4><<<gemmBlocks4, TPB, 0, stream>>>(xb, w1b, h, N, HD, FIN);
  k_node_scores<<<cdiv(N * HEADS, TPB), TPB, 0, stream>>>(h, a1s, a1d, es, ed, N, HEADS, HIDC, HD);
  k_fill_f32<<<cdiv(N * HEADS, TPB), TPB, 0, stream>>>(amax, NEG, N * HEADS);
  k_fill_f32<<<cdiv(N * HEADS, TPB), TPB, 0, stream>>>(den, 0.f, N * HEADS);
  k_fill_f32<<<cdiv(N * HD, TPB), TPB, 0, stream>>>(agg, 0.f, N * HD);
  k_edge_alpha_max<<<cdiv(EH, TPB), TPB, 0, stream>>>(ei, E, N, HEADS, es, ed, alpha, amax);
  k_edge_exp_sum  <<<cdiv(EH, TPB), TPB, 0, stream>>>(ei, E, N, HEADS, alpha, amax, den);
  k_edge_scatter  <<<cdiv(EH, TPB), TPB, 0, stream>>>(ei, E, N, HEADS, HIDC, HD, HD, alpha, den, h, agg);
  k_fill_f32<<<cdiv(HD, TPB), TPB, 0, stream>>>(bnsum, 0.f, HD);
  k_fill_f32<<<cdiv(HD, TPB), TPB, 0, stream>>>(bnsq, 0.f, HD);
  k_bn_stats<<<cdiv(N, 128), HD, 0, stream>>>(agg, b1, bnsum, bnsq, N, 128);
  k_bn_apply<<<cdiv(N * HD, TPB), TPB, 0, stream>>>(agg, b1, bnsum, bnsq, g1, be1, yb, N);

  // ========================= GAT layer 2 ==========================
  k_gemm_bf16<4><<<gemmBlocks4, TPB, 0, stream>>>(yb, w2b, h, N, HD, HD);
  k_node_scores<<<cdiv(N * HEADS, TPB), TPB, 0, stream>>>(h, a2s, a2d, es, ed, N, HEADS, HIDC, HD);
  k_fill_f32<<<cdiv(N * HEADS, TPB), TPB, 0, stream>>>(amax, NEG, N * HEADS);
  k_fill_f32<<<cdiv(N * HEADS, TPB), TPB, 0, stream>>>(den, 0.f, N * HEADS);
  k_fill_f32<<<cdiv(N * HD, TPB), TPB, 0, stream>>>(agg, 0.f, N * HD);
  k_edge_alpha_max<<<cdiv(EH, TPB), TPB, 0, stream>>>(ei, E, N, HEADS, es, ed, alpha, amax);
  k_edge_exp_sum  <<<cdiv(EH, TPB), TPB, 0, stream>>>(ei, E, N, HEADS, alpha, amax, den);
  k_edge_scatter  <<<cdiv(EH, TPB), TPB, 0, stream>>>(ei, E, N, HEADS, HIDC, HD, HD, alpha, den, h, agg);
  k_fill_f32<<<cdiv(HD, TPB), TPB, 0, stream>>>(bnsum, 0.f, HD);
  k_fill_f32<<<cdiv(HD, TPB), TPB, 0, stream>>>(bnsq, 0.f, HD);
  k_bn_stats<<<cdiv(N, 128), HD, 0, stream>>>(agg, b2, bnsum, bnsq, N, 128);
  k_bn_apply<<<cdiv(N * HD, TPB), TPB, 0, stream>>>(agg, b2, bnsum, bnsq, g2, be2, yb, N);

  // ========================= GAT layer 3 (H=1, C=10, padded to 16) ====
  k_gemm_bf16<1><<<cdiv(Mt, 8), TPB, 0, stream>>>(yb, w3b, h3, N, 16, HD);
  k_node_scores<<<cdiv(N, TPB), TPB, 0, stream>>>(h3, a3s, a3d, es, ed, N, 1, NCLS, 16);
  k_fill_f32<<<cdiv(N, TPB), TPB, 0, stream>>>(amax, NEG, N);
  k_fill_f32<<<cdiv(N, TPB), TPB, 0, stream>>>(den, 0.f, N);
  k_fill_f32<<<cdiv(N * NCLS, TPB), TPB, 0, stream>>>(out3, 0.f, N * NCLS);
  k_edge_alpha_max<<<cdiv(Ep, TPB), TPB, 0, stream>>>(ei, E, N, 1, es, ed, alpha, amax);
  k_edge_exp_sum  <<<cdiv(Ep, TPB), TPB, 0, stream>>>(ei, E, N, 1, alpha, amax, den);
  k_edge_scatter  <<<cdiv(Ep, TPB), TPB, 0, stream>>>(ei, E, N, 1, NCLS, 16, NCLS, alpha, den, h3, out3);

  // ========================= pooling + MLP head =======================
  k_fill_f32<<<1, TPB, 0, stream>>>(gsum, 0.f, NGRAPH * NCLS);
  k_fill_f32<<<1, TPB, 0, stream>>>(gmax, NEG, NGRAPH * NCLS);
  k_fill_f32<<<1, TPB, 0, stream>>>(gcnt, 0.f, NGRAPH);
  k_pool_acc<<<cdiv(N, TPB), TPB, 0, stream>>>(out3, b3, batch, gsum, gmax, gcnt, N);
  k_head_mlp<<<1, 32, 0, stream>>>(gsum, gmax, gcnt, Wp1, bp1, Wp2, bp2, (float*)d_out);
}
